// StackedTransEncoder_5085241279159
// MI455X (gfx1250) — matrix operand order repaired and checked by reading, not compiled, not run on hardware
//
#include <hip/hip_runtime.h>
#include <hip/hip_bf16.h>
#include <math.h>

#define Bb 64
#define Ll 512
#define Ee 512
#define Hh 512
#define G3 1536          // 3*H
#define LN_EPS 1e-5f

typedef __bf16 bf16_t;
typedef __attribute__((ext_vector_type(16))) __bf16 v16bf;
typedef __attribute__((ext_vector_type(8)))  __bf16 v8bf;
typedef __attribute__((ext_vector_type(8)))  float  v8f;

union V16 { v16bf v; v8bf h[2]; };

// ---------------------------------------------------------------------------
// WMMA helpers (CDNA5 wave32, D = A(16x32 bf16) * B(32x16 bf16) + C(16x16 f32))
// ---------------------------------------------------------------------------
__device__ __forceinline__ v8f wmma_bf16(v16bf a, v16bf b, v8f c) {
  return __builtin_amdgcn_wmma_f32_16x16x32_bf16(false, a, false, b,
                                                 (short)0, c, false, false);
}

// A fragment: 16x32 bf16 in this code's contiguous K order (not the ISA table order):
// lane l (l&15 = row M), l>>4 selects K-half: elements hold K=sel*16..sel*16+15
// (same K order as B). Row-major source, contiguous 16B loads.
__device__ __forceinline__ v16bf load_a_frag(const bf16_t* __restrict__ A, int lda,
                                             int row_base, int k0, int lane) {
  const int r   = row_base + (lane & 15);
  const int sel = lane >> 4;
  const bf16_t* p = A + (size_t)r * lda + k0 + sel * 16;
  V16 u;
  u.h[0] = *(const v8bf*)(p);
  u.h[1] = *(const v8bf*)(p + 8);
  return u.v;
}

// B fragment: 32x16 bf16, lane l holds column (l&15), K-half (l>>4)*16..+15.
// Source is pre-transposed (N-major, K contiguous) -> one 32B contiguous load.
__device__ __forceinline__ v16bf load_b_frag(const bf16_t* __restrict__ BT, int ldb,
                                             int col_base, int k0, int lane) {
  const int c  = col_base + (lane & 15);
  const int kh = lane >> 4;
  return *(const v16bf*)(BT + (size_t)c * ldb + k0 + kh * 16);
}

// ---------------------------------------------------------------------------
// Software-pipelined 16xK * Kx64 strip GEMM (one wave, 4 N-tiles, K=512).
// Fully unrolled (K is compile-time constant at both call sites) so the
// double-buffer rotation is resolved by SSA renaming (no v_dual_mov copies),
// and a sched_barrier pins the prefetch loads for k+32 ABOVE the WMMAs for k
// so the backend cannot sink them back to just-in-time position. The wait in
// front of each WMMA group then only covers loads issued a full iteration
// earlier (staggered s_wait_loadcnt instead of a 0x0 drain per WMMA).
// ---------------------------------------------------------------------------
__device__ __forceinline__ void gemm_strip_pipelined(
    const bf16_t* __restrict__ A, int lda, int row0,
    const bf16_t* __restrict__ BT, int ldb, int col0,
    int K, int lane, v8f acc[4]) {
  v16bf a_cur = load_a_frag(A, lda, row0, 0, lane);
  v16bf b_cur[4];
  #pragma unroll
  for (int i = 0; i < 4; ++i)
    b_cur[i] = load_b_frag(BT, ldb, col0 + i * 16, 0, lane);

  #pragma unroll
  for (int k0 = 0; k0 < K; k0 += 32) {
    const int kn = k0 + 32;
    v16bf a_nxt = a_cur;
    v16bf b_nxt[4];
    #pragma unroll
    for (int i = 0; i < 4; ++i) b_nxt[i] = b_cur[i];
    if (kn < K) {
      a_nxt = load_a_frag(A, lda, row0, kn, lane);
      #pragma unroll
      for (int i = 0; i < 4; ++i)
        b_nxt[i] = load_b_frag(BT, ldb, col0 + i * 16, kn, lane);
    }
    // Fence the scheduler: prefetch for k+32 stays above the WMMAs for k.
    __builtin_amdgcn_sched_barrier(0);
    #pragma unroll
    for (int i = 0; i < 4; ++i) acc[i] = wmma_bf16(a_cur, b_cur[i], acc[i]);
    a_cur = a_nxt;
    #pragma unroll
    for (int i = 0; i < 4; ++i) b_cur[i] = b_nxt[i];
  }
}

// ---------------------------------------------------------------------------
// Precision conversion / weight transpose
// ---------------------------------------------------------------------------
__global__ void convert_f32_bf16_kernel(const float* __restrict__ x,
                                        bf16_t* __restrict__ y, int n) {
  int i = blockIdx.x * blockDim.x + threadIdx.x;
  if (i < n) y[i] = (bf16_t)x[i];
}

// W: (K, G3) row-major fp32  ->  WT: (G3, K) row-major bf16
__global__ void transpose_w_kernel(const float* __restrict__ W,
                                   bf16_t* __restrict__ WT, int K) {
  int idx = blockIdx.x * blockDim.x + threadIdx.x;
  if (idx < K * G3) {
    int n = idx / K;
    int k = idx - n * K;
    WT[idx] = (bf16_t)W[(size_t)k * G3 + n];
  }
}

__global__ void init_state_kernel(float* __restrict__ h_f32,
                                  bf16_t* __restrict__ h_bf16,
                                  unsigned* __restrict__ bar, int nh) {
  int i = blockIdx.x * blockDim.x + threadIdx.x;
  if (i < nh) { h_f32[i] = 0.f; h_bf16[i] = (bf16_t)0.f; }
  if (i < 2) bar[i] = 0u;
}

// ---------------------------------------------------------------------------
// Kernel 1: px = LN3(xs @ Wx) for one direction (grid.z), bf16 out.
// Workgroup = 16 output rows x one 512-wide gate chunk; 8 waves x 4 N-tiles.
// LN reduction stays inside the workgroup via a 16x512 fp32 LDS tile.
// ---------------------------------------------------------------------------
__global__ __launch_bounds__(256) void px_gemm_ln_kernel(
    const bf16_t* __restrict__ xs,      // (B*L, E) bf16
    const bf16_t* __restrict__ WxT_f,   // (G3, E) bf16
    const bf16_t* __restrict__ WxT_b,
    const float* __restrict__ gx_f, const float* __restrict__ bx_f,
    const float* __restrict__ gx_b, const float* __restrict__ bx_b,
    bf16_t* __restrict__ px)            // (2, B*L, G3) bf16 (post-LN)
{
  __shared__ float tile[16 * 512];
  __shared__ float red_s[16][17];
  __shared__ float red_q[16][17];
  __shared__ float s_mu[16], s_rs[16];

  const int rowblk = blockIdx.x;        // 0..2047 (16-row blocks of B*L)
  const int gate   = blockIdx.y;        // 0..2
  const int dir    = blockIdx.z;        // 0..1
  const int tid  = threadIdx.x;
  const int lane = tid & 31;
  const int wave = tid >> 5;

  const bf16_t* WT = dir ? WxT_b : WxT_f;
  const float*  gg = dir ? gx_b : gx_f;
  const float*  bc = dir ? bx_b : bx_f;

  const int row0  = rowblk * 16;
  const int colg0 = gate * 512;
  const int col0  = colg0 + wave * 64;  // this wave's 4 N-tiles

  v8f acc[4];
  #pragma unroll
  for (int i = 0; i < 4; ++i) acc[i] = v8f{};

  gemm_strip_pipelined(xs, Ee, row0, WT, Ee, col0, Ee, lane, acc);

  // C/D layout: VGPR r -> lanes0-15: (M=r, N=lane); lanes16-31: (M=8+r, N=lane-16)
  const int m0  = (lane < 16) ? 0 : 8;
  const int nlc = lane & 15;
  #pragma unroll
  for (int i = 0; i < 4; ++i) {
    int c = wave * 64 + i * 16 + nlc;
    #pragma unroll
    for (int r = 0; r < 8; ++r) tile[(m0 + r) * 512 + c] = acc[i][r];
  }
  __syncthreads();

  // LayerNorm over the 512-wide gate chunk, per row
  const int r  = tid >> 4;   // 0..15
  const int cb = tid & 15;
  float s = 0.f, q = 0.f;
  for (int j = 0; j < 32; ++j) {
    float v = tile[r * 512 + cb + j * 16];
    s += v; q += v * v;
  }
  red_s[r][cb] = s; red_q[r][cb] = q;
  __syncthreads();
  if (cb == 0) {
    float S = 0.f, Q = 0.f;
    for (int j = 0; j < 16; ++j) { S += red_s[r][j]; Q += red_q[r][j]; }
    float mu  = S * (1.f / 512.f);
    float var = Q * (1.f / 512.f) - mu * mu;
    s_mu[r] = mu;
    s_rs[r] = rsqrtf(var + LN_EPS);
  }
  __syncthreads();

  bf16_t* out = px + (size_t)dir * (size_t)(Bb * Ll) * G3;
  for (int j = 0; j < 32; ++j) {
    int c  = cb + j * 16;
    int gc = colg0 + c;
    float y = (tile[r * 512 + c] - s_mu[r]) * s_rs[r] * gg[gc] + bc[gc];
    out[(size_t)(row0 + r) * G3 + gc] = (bf16_t)y;
  }
}

// ---------------------------------------------------------------------------
// Device-wide barrier (generation counter; graph-capture safe, no host sync)
// ---------------------------------------------------------------------------
__device__ __forceinline__ void grid_barrier(unsigned* bar, unsigned nblocks) {
  __syncthreads();
  if (threadIdx.x == 0) {
    __threadfence();
    unsigned gen  = atomicAdd(&bar[1], 0u);
    unsigned prev = atomicAdd(&bar[0], 1u);
    if (prev == nblocks - 1u) {
      atomicExch(&bar[0], 0u);
      __threadfence();
      atomicAdd(&bar[1], 1u);
    } else {
      while (atomicAdd(&bar[1], 0u) == gen) { __builtin_amdgcn_s_sleep(2); }
    }
  }
  __syncthreads();
}

// ---------------------------------------------------------------------------
// Kernel 2: persistent bidirectional GRU scan. 24 workgroups:
//   block = (dir in 2) x (mtile in 4: 16 batch rows) x (gate in 3)
// Per step: phase1 GEMM h@WhT + LN (workgroup-local) -> ph fp32;
//           grid barrier; phase2 (gate==0 blocks) gate-combine + h update;
//           grid barrier.
// Wh (bf16, 1.5MB/dir) stays hot in the 192MB L2 across all 512 steps.
// ---------------------------------------------------------------------------
__global__ __launch_bounds__(256) void gru_scan_kernel(
    const bf16_t* __restrict__ px,      // (2, B*L, G3) post-LN bf16
    const float*  __restrict__ mask,    // (B, L)
    const bf16_t* __restrict__ WhT_f,   // (G3, H) bf16
    const bf16_t* __restrict__ WhT_b,
    const float* __restrict__ gh_f, const float* __restrict__ bh_f,
    const float* __restrict__ gh_b, const float* __restrict__ bh_b,
    float*  __restrict__ h_f32,         // (2, B, H)
    bf16_t* __restrict__ h_bf16,        // (2, B, H)
    float*  __restrict__ ph,            // (2, B, G3)
    float*  __restrict__ out,           // (B, L, 2H)
    unsigned* __restrict__ bar)
{
  __shared__ float tile[16 * 512];
  __shared__ float red_s[16][17];
  __shared__ float red_q[16][17];
  __shared__ float s_mu[16], s_rs[16];

  const int bi    = blockIdx.x;         // 0..23
  const int dir   = bi / 12;
  const int rem   = bi - dir * 12;
  const int mtile = rem / 3;            // 0..3 -> batch rows mtile*16..+15
  const int gate  = rem - mtile * 3;    // 0..2
  const unsigned NB = 24u;

  const int tid  = threadIdx.x;
  const int lane = tid & 31;
  const int wave = tid >> 5;

  const bf16_t* WT = dir ? WhT_b : WhT_f;
  const float*  gg = dir ? gh_b : gh_f;
  const float*  bc = dir ? bh_b : bh_f;

  const bf16_t* hA    = h_bf16 + (size_t)dir * Bb * Hh;
  float*        phD   = ph + (size_t)dir * Bb * G3;
  const int     row0  = mtile * 16;
  const int     colg0 = gate * 512;
  const int     col0  = colg0 + wave * 64;

  for (int t = 0; t < Ll; ++t) {
    // ---- phase 1: ph_chunk = LN3(h @ Wh)[rows, gate cols] ----
    v8f acc[4];
    #pragma unroll
    for (int i = 0; i < 4; ++i) acc[i] = v8f{};

    gemm_strip_pipelined(hA, Hh, row0, WT, Hh, col0, Hh, lane, acc);

    const int m0  = (lane < 16) ? 0 : 8;
    const int nlc = lane & 15;
    #pragma unroll
    for (int i = 0; i < 4; ++i) {
      int c = wave * 64 + i * 16 + nlc;
      #pragma unroll
      for (int r = 0; r < 8; ++r) tile[(m0 + r) * 512 + c] = acc[i][r];
    }
    __syncthreads();

    const int r  = tid >> 4;
    const int cb = tid & 15;
    float s = 0.f, q = 0.f;
    for (int j = 0; j < 32; ++j) {
      float v = tile[r * 512 + cb + j * 16];
      s += v; q += v * v;
    }
    red_s[r][cb] = s; red_q[r][cb] = q;
    __syncthreads();
    if (cb == 0) {
      float S = 0.f, Q = 0.f;
      for (int j = 0; j < 16; ++j) { S += red_s[r][j]; Q += red_q[r][j]; }
      float mu  = S * (1.f / 512.f);
      float var = Q * (1.f / 512.f) - mu * mu;
      s_mu[r] = mu;
      s_rs[r] = rsqrtf(var + LN_EPS);
    }
    __syncthreads();
    for (int j = 0; j < 32; ++j) {
      int c  = cb + j * 16;
      int gc = colg0 + c;
      float y = (tile[r * 512 + c] - s_mu[r]) * s_rs[r] * gg[gc] + bc[gc];
      phD[(size_t)(row0 + r) * G3 + gc] = y;
    }

    grid_barrier(bar, NB);

    // ---- phase 2: combine gates, update h, write output ----
    if (gate == 0) {
      const int l = dir ? (Ll - 1 - t) : t;
      for (int idx = tid; idx < 16 * 512; idx += 256) {
        int rr = idx >> 9;
        int c  = idx & 511;
        int b  = row0 + rr;
        size_t prow = ((size_t)dir * (Bb * Ll) + (size_t)b * Ll + l) * G3;
        float xz = (float)px[prow + c];
        float xr = (float)px[prow + 512 + c];
        float xg = (float)px[prow + 1024 + c];
        size_t hrow = (size_t)b * G3;
        float hz = phD[hrow + c];
        float hr = phD[hrow + 512 + c];
        float hg = phD[hrow + 1024 + c];
        float z  = 1.f / (1.f + __expf(-(xz + hz)));
        float rg = 1.f / (1.f + __expf(-(xr + hr)));
        float g  = tanhf(xg + rg * hg);
        size_t hidx = (size_t)dir * Bb * Hh + (size_t)b * Hh + c;
        float hp = h_f32[hidx];
        float hn = (1.f - z) * hp + z * g;
        float m  = mask[(size_t)b * Ll + l];
        float hc = m * hn + (1.f - m) * hp;
        h_f32[hidx]  = hc;
        h_bf16[hidx] = (bf16_t)hc;
        out[((size_t)b * Ll + l) * (2 * Hh) + dir * Hh + c] = hc * m;
      }
    }

    grid_barrier(bar, NB);
  }
}

// ---------------------------------------------------------------------------
extern "C" void kernel_launch(void* const* d_in, const int* in_sizes, int n_in,
                              void* d_out, int out_size, void* d_ws, size_t ws_size,
                              hipStream_t stream) {
  const float* xs_e    = (const float*)d_in[0];
  const float* xs_mask = (const float*)d_in[1];
  const float* f_Wx = (const float*)d_in[2];
  const float* f_Wh = (const float*)d_in[3];
  const float* f_gx = (const float*)d_in[4];
  const float* f_bx = (const float*)d_in[5];
  const float* f_gh = (const float*)d_in[6];
  const float* f_bh = (const float*)d_in[7];
  const float* b_Wx = (const float*)d_in[8];
  const float* b_Wh = (const float*)d_in[9];
  const float* b_gx = (const float*)d_in[10];
  const float* b_bx = (const float*)d_in[11];
  const float* b_gh = (const float*)d_in[12];
  const float* b_bh = (const float*)d_in[13];

  // Workspace carve-up (256B aligned)
  char* w = (char*)d_ws;
  auto alloc = [&](size_t bytes) -> void* {
    void* p = (void*)w;
    w += (bytes + 255) & ~(size_t)255;
    return p;
  };
  const size_t BLE = (size_t)Bb * Ll * Ee;
  bf16_t*  xs_bf   = (bf16_t*)alloc(BLE * 2);
  bf16_t*  WxT_f   = (bf16_t*)alloc((size_t)G3 * Ee * 2);
  bf16_t*  WxT_b   = (bf16_t*)alloc((size_t)G3 * Ee * 2);
  bf16_t*  WhT_f   = (bf16_t*)alloc((size_t)G3 * Hh * 2);
  bf16_t*  WhT_b   = (bf16_t*)alloc((size_t)G3 * Hh * 2);
  bf16_t*  px      = (bf16_t*)alloc((size_t)2 * Bb * Ll * G3 * 2);
  float*   ph      = (float*)alloc((size_t)2 * Bb * G3 * 4);
  float*   h_f32   = (float*)alloc((size_t)2 * Bb * Hh * 4);
  bf16_t*  h_bf16  = (bf16_t*)alloc((size_t)2 * Bb * Hh * 2);
  unsigned* bar    = (unsigned*)alloc(256);

  // 1) precision conversion + weight transpose (bf16, N-major for B-fragments)
  convert_f32_bf16_kernel<<<(int)((BLE + 255) / 256), 256, 0, stream>>>(
      xs_e, xs_bf, (int)BLE);
  const int nW = Ee * G3;
  transpose_w_kernel<<<(nW + 255) / 256, 256, 0, stream>>>(f_Wx, WxT_f, Ee);
  transpose_w_kernel<<<(nW + 255) / 256, 256, 0, stream>>>(b_Wx, WxT_b, Ee);
  transpose_w_kernel<<<(nW + 255) / 256, 256, 0, stream>>>(f_Wh, WhT_f, Hh);
  transpose_w_kernel<<<(nW + 255) / 256, 256, 0, stream>>>(b_Wh, WhT_b, Hh);

  // 2) big parallel GEMM + fused LayerNorm for both directions
  dim3 pg(Bb * Ll / 16, 3, 2);   // (2048, 3, 2) workgroups of 8 waves
  px_gemm_ln_kernel<<<pg, 256, 0, stream>>>(xs_bf, WxT_f, WxT_b,
                                            f_gx, f_bx, b_gx, b_bx, px);

  // 3) state + barrier init (ws is poisoned; must re-init every call)
  init_state_kernel<<<(2 * Bb * Hh + 255) / 256, 256, 0, stream>>>(
      h_f32, h_bf16, bar, 2 * Bb * Hh);

  // 4) persistent recurrent scan (both directions concurrently)
  gru_scan_kernel<<<24, 256, 0, stream>>>(px, xs_mask, WhT_f, WhT_b,
                                          f_gh, f_bh, b_gh, b_bh,
                                          h_f32, h_bf16, ph, (float*)d_out, bar);
}
